// GCN_83545703842127
// MI455X (gfx1250) — matrix operand order repaired and checked by reading, not compiled, not run on hardware
//
#include <hip/hip_runtime.h>
#include <hip/hip_bf16.h>

// ---- CDNA5 WMMA types ------------------------------------------------------
typedef __attribute__((ext_vector_type(16))) __bf16 v16bf;
typedef __attribute__((ext_vector_type(8)))  __bf16 bf16x8;
typedef __attribute__((ext_vector_type(8)))  float  v8f;

union Frag16 {            // 32 bytes: one WMMA A/B bf16 operand per lane
    v16bf  v;
    bf16x8 h[2];
};

// ---- small elementwise kernels --------------------------------------------
__global__ void k_fill_deg(float* deg, int n) {
    int i = blockIdx.x * blockDim.x + threadIdx.x;
    if (i < n) deg[i] = 1.0f;                 // self-loop contributes 1 to deg
}

__global__ void k_deg_edges(const int* __restrict__ col, float* deg, int E) {
    int e = blockIdx.x * blockDim.x + threadIdx.x;
    if (e < E) atomicAdd(&deg[col[e]], 1.0f);
}

__global__ void k_dinv(const float* __restrict__ deg, float* dinv, int n) {
    int i = blockIdx.x * blockDim.x + threadIdx.x;
    if (i < n) {
        float d = deg[i];
        dinv[i] = d > 0.0f ? rsqrtf(d) : 0.0f;
    }
}

// f32 -> bf16 hi/lo split (hi + lo reproduces ~16 mantissa bits)
__global__ void k_split_bf16(const float* __restrict__ x, __bf16* __restrict__ hi,
                             __bf16* __restrict__ lo, size_t n) {
    size_t i = (size_t)blockIdx.x * blockDim.x + threadIdx.x;
    if (i >= n) return;
    float v = x[i];
    __bf16 h = (__bf16)v;
    hi[i] = h;
    lo[i] = (__bf16)(v - (float)h);
}

// Pack W[K x N] (row-major f32) into per-lane WMMA B fragments, bf16 hi/lo.
// B 32x16 bf16 layout (wave32): lane col = lane&15; K group = (lane>>4)*8;
// half i of the lane's 16 halves holds B[kt*32 + kgrp + (i&7) + (i>>3)*16][nt*16 + col].
__global__ void k_pack_w(const float* __restrict__ W, __bf16* __restrict__ outH,
                         __bf16* __restrict__ outL, int K, int N) {
    int NT = N / 16, KT = K / 32;
    int tid = blockIdx.x * blockDim.x + threadIdx.x;
    if (tid >= KT * NT * 32) return;
    int lane = tid & 31;
    int frag = tid >> 5;
    int kt = frag / NT;
    int nt = frag - kt * NT;
    int n     = nt * 16 + (lane & 15);
    int kbase = kt * 32 + ((lane >> 4) * 8);
    size_t o = (size_t)tid * 16;              // [kt][nt][lane][16 halves]
    for (int i = 0; i < 16; ++i) {
        float v  = W[(size_t)(kbase + (i & 7) + ((i >> 3) << 4)) * N + n];
        __bf16 h = (__bf16)v;
        outH[o + i] = h;
        outL[o + i] = (__bf16)(v - (float)h);
    }
}

// ---- WMMA GEMM: out[M x N] = (Ahi+Alo) @ W, bf16x3 emulation of fp32 -------
// One wave computes a 16-row strip x NTW 16-col tiles; 3 bf16 WMMAs per k-tile
// (hi*hi + hi*lo + lo*hi) into the f32 accumulator. All addressing in the hot
// loop is pointer-increment + immediate-offset so B loads become
// global_load_b128 base, off:<const> with one u64 add per operand per k-tile.
// Requires M%16==0, K%32==0, N%(16*NTW)==0; wave-uniform exit keeps EXEC all-1.
template <int NTW>
__global__ void k_gemm_wmma_bf16x3(const __bf16* __restrict__ aH,
                                   const __bf16* __restrict__ aL,
                                   const __bf16* __restrict__ bH,
                                   const __bf16* __restrict__ bL,
                                   float* __restrict__ out,
                                   int K, int N, int totalWaves) {
    int wave = blockIdx.x * (blockDim.x >> 5) + (threadIdx.x >> 5);
    if (wave >= totalWaves) return;           // uniform across the wave
    int lane    = threadIdx.x & 31;
    int ngroups = N / (16 * NTW);
    int mg = wave / ngroups;
    int ng = wave - mg * ngroups;
    int m0 = mg * 16;
    int NTtot  = N / 16;
    int ntbase = ng * NTW;

    v8f acc[NTW] = {};

    // A: lane row = m0+(lane&15); halves 0..7 at k = kt*32 + (lane>>4)*8,
    // halves 8..15 at +16 (16-bit A 16x32 layout).
    size_t arow = (size_t)(m0 + (lane & 15)) * K + ((lane >> 4) * 8);
    const __bf16* pAH = aH + arow;
    const __bf16* pAL = aL + arow;
    // B: this wave's first fragment for kt=0; per-nt offset is the constant
    // nt*512 halves (1024 B), per-kt stride is NTtot*512 halves.
    size_t bofs = ((size_t)ntbase * 32 + lane) * 16;
    const __bf16* pBH = bH + bofs;
    const __bf16* pBL = bL + bofs;
    const size_t  bKtStride = (size_t)NTtot * 512;

    for (int kt = 0; kt < (K >> 5); ++kt) {
        Frag16 ah, al;
        ah.h[0] = *(const bf16x8*)(pAH);
        ah.h[1] = *(const bf16x8*)(pAH + 16);
        al.h[0] = *(const bf16x8*)(pAL);
        al.h[1] = *(const bf16x8*)(pAL + 16);
        pAH += 32;
        pAL += 32;
#pragma unroll
        for (int nt = 0; nt < NTW; ++nt) {
            Frag16 bh, bl;
            bh.h[0] = *(const bf16x8*)(pBH + nt * 512);      // immediate offsets
            bh.h[1] = *(const bf16x8*)(pBH + nt * 512 + 8);
            bl.h[0] = *(const bf16x8*)(pBL + nt * 512);
            bl.h[1] = *(const bf16x8*)(pBL + nt * 512 + 8);
            acc[nt] = __builtin_amdgcn_wmma_f32_16x16x32_bf16(
                false, ah.v, false, bh.v, (short)0, acc[nt], false, false);
            acc[nt] = __builtin_amdgcn_wmma_f32_16x16x32_bf16(
                false, ah.v, false, bl.v, (short)0, acc[nt], false, false);
            acc[nt] = __builtin_amdgcn_wmma_f32_16x16x32_bf16(
                false, al.v, false, bh.v, (short)0, acc[nt], false, false);
        }
        pBH += bKtStride;
        pBL += bKtStride;
    }

    // C/D 16x16 f32 layout: VGPR r -> row m0 + r + (lane>>4)*8, col = lane&15.
    int ms = m0 + ((lane >> 4) * 8);
#pragma unroll
    for (int nt = 0; nt < NTW; ++nt) {
        int ncol = (ntbase + nt) * 16 + (lane & 15);
        float* o = out + (size_t)ms * N + ncol;
#pragma unroll
        for (int r = 0; r < 8; ++r) o[(size_t)r * N] = acc[nt][r];
    }
}

// agg[i,f] = dinv[i]^2 * h[i,f]   (self-loop term, also initializes buffer)
__global__ void k_self_init(const float* __restrict__ h, const float* __restrict__ dinv,
                            float* __restrict__ agg, int n, int F) {
    size_t i = (size_t)blockIdx.x * blockDim.x + threadIdx.x;
    if (i >= (size_t)n * F) return;
    float d = dinv[i / F];
    agg[i] = d * d * h[i];
}

// Edge scatter: 32 edges per wave; per edge, norm broadcast via shuffles,
// lanes stream F features with coalesced loads + global f32 atomics (L2-resident).
__global__ void k_scatter(const int* __restrict__ rows, const int* __restrict__ cols,
                          const float* __restrict__ dinv, const float* __restrict__ h,
                          float* __restrict__ agg, int E, int F) {
    int lane = threadIdx.x & 31;
    int wave = blockIdx.x * (blockDim.x >> 5) + (threadIdx.x >> 5);
    int e0 = wave * 32;
    if (e0 >= E) return;
    int r = 0, c = 0; float nrm = 0.0f;
    int e = e0 + lane;
    if (e < E) {
        r = rows[e];
        c = cols[e];
        nrm = dinv[r] * dinv[c];
    }
    int cnt = (E - e0 < 32) ? (E - e0) : 32;
    for (int j = 0; j < cnt; ++j) {
        int   rj = __shfl(r, j);
        int   cj = __shfl(c, j);
        float nj = __shfl(nrm, j);
        const float* hr = h   + (size_t)rj * F;
        float*       ag = agg + (size_t)cj * F;
        for (int f = lane; f < F; f += 32) atomicAdd(ag + f, nj * hr[f]);
    }
}

__global__ void k_bias_act(float* __restrict__ a, const float* __restrict__ b,
                           size_t n, int F, int doRelu) {
    size_t i = (size_t)blockIdx.x * blockDim.x + threadIdx.x;
    if (i >= n) return;
    float v = a[i] + b[i % F];
    a[i] = doRelu ? fmaxf(v, 0.0f) : v;
}

// ---------------------------------------------------------------------------
static inline unsigned cdiv(size_t a, unsigned b) { return (unsigned)((a + b - 1) / b); }

extern "C" void kernel_launch(void* const* d_in, const int* in_sizes, int n_in,
                              void* d_out, int out_size, void* d_ws, size_t ws_size,
                              hipStream_t stream) {
    const float* x    = (const float*)d_in[0];
    const int*   edge = (const int*)d_in[1];
    const float* W1   = (const float*)d_in[2];
    const float* b1   = (const float*)d_in[3];
    const float* W2   = (const float*)d_in[4];
    const float* b2   = (const float*)d_in[5];

    const int H = in_sizes[3];                // 256
    const int C = in_sizes[5];                // 64
    const int F = in_sizes[2] / H;            // 512
    const int N = in_sizes[0] / F;            // 50000
    const int E = in_sizes[1] / 2;            // 800000
    const int* rows = edge;
    const int* cols = edge + E;

    // ---- workspace layout (bump allocator, 256B aligned) ----
    char* ws = (char*)d_ws;
    size_t off = 0;
    auto bump = [&](size_t bytes) {
        size_t o = off;
        off += (bytes + 255) & ~(size_t)255;
        return (void*)(ws + o);
    };
    float*  deg  = (float*) bump((size_t)N * 4);
    float*  dinv = (float*) bump((size_t)N * 4);
    __bf16* w1h  = (__bf16*)bump((size_t)F * H * 2);
    __bf16* w1l  = (__bf16*)bump((size_t)F * H * 2);
    __bf16* w2h  = (__bf16*)bump((size_t)H * C * 2);
    __bf16* w2l  = (__bf16*)bump((size_t)H * C * 2);
    __bf16* xh   = (__bf16*)bump((size_t)N * F * 2);   // reused as y_hi/y_lo later
    __bf16* xl   = (__bf16*)bump((size_t)N * F * 2);
    float*  h1   = (float*) bump((size_t)N * H * 4);   // reused as h2 later
    float*  agg1 = (float*) bump((size_t)N * H * 4);
    __bf16* yh   = xh;                                 // N*H bf16
    __bf16* yl   = xh + (size_t)N * H;                 // N*H bf16 (within x region)
    float*  h2   = h1;                                 // N*C f32 (h1 dead by then)
    float*  outp = (float*)d_out;                      // N*C accumulation target

    const unsigned TB = 256;

    // 1) degree + rsqrt norm
    k_fill_deg <<<cdiv(N, TB), TB, 0, stream>>>(deg, N);
    k_deg_edges<<<cdiv(E, TB), TB, 0, stream>>>(cols, deg, E);
    k_dinv     <<<cdiv(N, TB), TB, 0, stream>>>(deg, dinv, N);

    // 2) split x, pack W1
    k_split_bf16<<<cdiv((size_t)N * F, TB), TB, 0, stream>>>(x, xh, xl, (size_t)N * F);
    k_pack_w   <<<cdiv((size_t)(F / 32) * (H / 16) * 32, TB), TB, 0, stream>>>(W1, w1h, w1l, F, H);

    // 3) GEMM1: h1 = x @ W1  (NTW=8 -> 128 cols per wave, 2 n-groups for H=256)
    {
        int waves = (N / 16) * (H / (16 * 8));
        k_gemm_wmma_bf16x3<8><<<cdiv((size_t)waves * 32, TB), TB, 0, stream>>>(
            xh, xl, w1h, w1l, h1, F, H, waves);
    }

    // 4) aggregate layer 1: agg1 = selfloop + sum_e norm*h1[row], then relu(+b1)
    k_self_init<<<cdiv((size_t)N * H, TB), TB, 0, stream>>>(h1, dinv, agg1, N, H);
    k_scatter  <<<cdiv((size_t)E, 32 * (TB / 32)) , TB, 0, stream>>>(rows, cols, dinv, h1, agg1, E, H);
    k_bias_act <<<cdiv((size_t)N * H, TB), TB, 0, stream>>>(agg1, b1, (size_t)N * H, H, 1);

    // 5) split y1, pack W2, GEMM2: h2 = y1 @ W2  (NTW=4 -> 64 cols per wave)
    k_split_bf16<<<cdiv((size_t)N * H, TB), TB, 0, stream>>>(agg1, yh, yl, (size_t)N * H);
    k_pack_w    <<<cdiv((size_t)(H / 32) * (C / 16) * 32, TB), TB, 0, stream>>>(W2, w2h, w2l, H, C);
    {
        int waves = (N / 16) * (C / (16 * 4));
        k_gemm_wmma_bf16x3<4><<<cdiv((size_t)waves * 32, TB), TB, 0, stream>>>(
            yh, yl, w2h, w2l, h2, H, C, waves);
    }

    // 6) aggregate layer 2 directly into d_out, add bias (no relu)
    k_self_init<<<cdiv((size_t)N * C, TB), TB, 0, stream>>>(h2, dinv, outp, N, C);
    k_scatter  <<<cdiv((size_t)E, 32 * (TB / 32)), TB, 0, stream>>>(rows, cols, dinv, h2, outp, E, C);
    k_bias_act <<<cdiv((size_t)N * C, TB), TB, 0, stream>>>(outp, b2, (size_t)N * C, C, 0);
}
